// FluxIPAttnProcessor_41618233098596
// MI455X (gfx1250) — matrix operand order repaired and checked by reading, not compiled, not run on hardware
//
#include <hip/hip_runtime.h>
#include <hip/hip_bf16.h>

// ---------------- problem constants ----------------
#define H_     24
#define D_     128
#define HID_   3072
#define S_TXT_ 512
#define S_IMG_ 1024
#define S_ALL_ 1536
#define L_IP_  16
#define KVP_IP 32          // IP kv padded to one 32-key WMMA tile
#define IPD_   2048
#define EPS_   1e-6f

typedef __attribute__((ext_vector_type(16))) __bf16 v16bf;
typedef __attribute__((ext_vector_type(8)))  float  v8f;

union F16x16 { uint4 u[2]; __bf16 b[16]; v16bf v; };
union F16x4  { uint2 u;    __bf16 b[4]; };

static __device__ __forceinline__ v8f v8f_zero() {
  v8f z = {0.f,0.f,0.f,0.f,0.f,0.f,0.f,0.f};
  return z;
}
static __device__ __forceinline__ float red_max16(float v) {
#pragma unroll
  for (int m = 1; m < 16; m <<= 1) v = fmaxf(v, __shfl_xor(v, m, 32));
  return v;
}
static __device__ __forceinline__ float red_sum16(float v) {
#pragma unroll
  for (int m = 1; m < 16; m <<= 1) v += __shfl_xor(v, m, 32);
  return v;
}
static __device__ __forceinline__ float red_sum32(float v) {
#pragma unroll
  for (int m = 1; m < 32; m <<= 1) v += __shfl_xor(v, m, 32);
  return v;
}

// Async global->LDS copy of 64 contiguous bytes (4 x b128).
// INST_OFFSET advances both the LDS and global addresses (ISA 10.x / 15.18.3).
// LDS byte address = low 32 bits of the generic pointer (ISA flat aperture rule).
static __device__ __forceinline__ void async_copy64(const __bf16* gsrc, __bf16* ldst) {
  unsigned lds_off = (unsigned)(unsigned long long)ldst;
  asm volatile(
      "global_load_async_to_lds_b128 %0, %1, off\n\t"
      "global_load_async_to_lds_b128 %0, %1, off offset:16\n\t"
      "global_load_async_to_lds_b128 %0, %1, off offset:32\n\t"
      "global_load_async_to_lds_b128 %0, %1, off offset:48"
      :: "v"(lds_off), "v"(gsrc) : "memory");
}
static __device__ __forceinline__ void async_wait0() {
  asm volatile("s_wait_asynccnt 0x0" ::: "memory");
}

// =====================================================================
// GEMM:  C[M,N] f32 = (A + A2?)[M,K] f32  @  W[K,N] f32  + bias[N]
// inline f32->bf16 while staging to LDS; WMMA bf16, f32 acc.
// ldsA row-major [m][k] pitch 40; ldsB column-major [n][k] pitch 40
// so every WMMA fragment is two contiguous 16B LDS loads.
// =====================================================================
#define BM 128
#define BN 128
#define BK 32
#define APITCH 40
#define BPITCH 40

__global__ __launch_bounds__(256) void gemm_bias_kernel(
    const float* __restrict__ A, const float* __restrict__ A2,
    const float* __restrict__ W, const float* __restrict__ bias,
    float* __restrict__ C, int M, int N, int K)
{
  __shared__ __align__(16) __bf16 ldsA[BM][APITCH];
  __shared__ __align__(16) __bf16 ldsB[BN][BPITCH];   // [n][k] column-major

  const int tid  = threadIdx.x;
  const int wave = tid >> 5, lane = tid & 31;
  const int half = lane >> 4, l16 = lane & 15;
  const int wm = wave >> 1, wn = wave & 1;            // 4x2 wave grid
  const int mtile = blockIdx.y * BM, ntile = blockIdx.x * BN;

  v8f acc[2][4];
#pragma unroll
  for (int i = 0; i < 2; ++i)
#pragma unroll
    for (int j = 0; j < 4; ++j) acc[i][j] = v8f_zero();

  const int arow = tid >> 1, acol = (tid & 1) * 16;      // A: 128 rows x 32 cols
  const int kb4 = (tid >> 5) * 4, nb4 = (tid & 31) * 4;  // W: 8x32 blocks of 4x4
  const bool hasA2 = (A2 != nullptr);
  const bool arowOk = (mtile + arow) < M;

  for (int k0 = 0; k0 < K; k0 += BK) {
    __syncthreads();
    // ---- prefetch next tiles (global_prefetch_b8) ----
    if (k0 + BK < K) {
      if (arowOk)
        __builtin_prefetch(A + (size_t)(mtile + arow) * K + k0 + BK + acol, 0, 1);
      __builtin_prefetch(W + (size_t)(k0 + BK + kb4) * N + ntile + nb4, 0, 1);
    }
    // ---- stage A tile (f32 -> bf16) ----
    {
      F16x16 st;
      if (arowOk) {
        const float4* ap = reinterpret_cast<const float4*>(A + (size_t)(mtile + arow) * K + k0 + acol);
        float t[16];
#pragma unroll
        for (int i = 0; i < 4; ++i) reinterpret_cast<float4*>(t)[i] = ap[i];
        if (hasA2) {
          const float4* ap2 = reinterpret_cast<const float4*>(A2 + (size_t)(mtile + arow) * K + k0 + acol);
#pragma unroll
          for (int i = 0; i < 4; ++i) {
            float4 u = ap2[i];
            t[4*i+0] += u.x; t[4*i+1] += u.y; t[4*i+2] += u.z; t[4*i+3] += u.w;
          }
        }
#pragma unroll
        for (int j = 0; j < 16; ++j) st.b[j] = (__bf16)t[j];
      } else {
#pragma unroll
        for (int j = 0; j < 16; ++j) st.b[j] = (__bf16)0.f;
      }
      *reinterpret_cast<uint4*>(&ldsA[arow][acol])     = st.u[0];
      *reinterpret_cast<uint4*>(&ldsA[arow][acol + 8]) = st.u[1];
    }
    // ---- stage W tile into column-major [n][k] ----
    {
      float t[4][4];
#pragma unroll
      for (int r = 0; r < 4; ++r)
        reinterpret_cast<float4*>(t[r])[0] =
            *reinterpret_cast<const float4*>(W + (size_t)(k0 + kb4 + r) * N + ntile + nb4);
#pragma unroll
      for (int j = 0; j < 4; ++j) {
        F16x4 st;
#pragma unroll
        for (int r = 0; r < 4; ++r) st.b[r] = (__bf16)t[r][j];
        *reinterpret_cast<uint2*>(&ldsB[nb4 + j][kb4]) = st.u;
      }
    }
    __syncthreads();

    // ---- fragments (2x b128 each) + 8 WMMAs ----
    v16bf afrag[2], bfrag[4];
#pragma unroll
    for (int fm = 0; fm < 2; ++fm) {
      const int row = wm * 32 + fm * 16 + l16;
      F16x16 f;
      f.u[0] = *reinterpret_cast<const uint4*>(&ldsA[row][half * 8]);
      f.u[1] = *reinterpret_cast<const uint4*>(&ldsA[row][16 + half * 8]);
      afrag[fm] = f.v;
    }
#pragma unroll
    for (int fn = 0; fn < 4; ++fn) {
      const int col = wn * 64 + fn * 16 + l16;
      F16x16 f;
      f.u[0] = *reinterpret_cast<const uint4*>(&ldsB[col][half * 8]);
      f.u[1] = *reinterpret_cast<const uint4*>(&ldsB[col][16 + half * 8]);
      bfrag[fn] = f.v;
    }
#pragma unroll
    for (int fm = 0; fm < 2; ++fm)
#pragma unroll
      for (int fn = 0; fn < 4; ++fn)
        acc[fm][fn] = __builtin_amdgcn_wmma_f32_16x16x32_bf16(
            false, afrag[fm], false, bfrag[fn], (short)0, acc[fm][fn], false, false);
  }

  // ---- epilogue: bias + store ----
#pragma unroll
  for (int fm = 0; fm < 2; ++fm) {
#pragma unroll
    for (int fn = 0; fn < 4; ++fn) {
      const int n  = ntile + wn * 64 + fn * 16 + l16;
      const float bv = bias ? bias[n] : 0.f;
#pragma unroll
      for (int r = 0; r < 8; ++r) {
        const int m = mtile + wm * 32 + fm * 16 + half * 8 + r;
        if (m < M) C[(size_t)m * N + n] = acc[fm][fn][r] + bv;
      }
    }
  }
}

// =====================================================================
// Flash attention.  Q:[H][Sq][128] bf16, K:[H][SkvPad][128] bf16,
// VT:[H][128][SkvPad] bf16.  Out f32 [Sq][HID_] at column h*128.
// Block = 128 thr (4 waves); each wave owns 16 q-rows; 32-key KV tiles.
// KV tiles staged with GLOBAL_LOAD_ASYNC_TO_LDS_B128 (ASYNCcnt).
// =====================================================================
#define FQ_TILE 64
#define FKV     32
#define KPITCH  136   // ldsK  [key][d]   : 272B rows (16B aligned)
#define VPITCH  40    // ldsVT [d][key]   : 80B rows
#define PPITCH  40    // ldsP  [qrow][key]: 80B rows

__global__ __launch_bounds__(128) void flash_attn_kernel(
    const __bf16* __restrict__ Q, const __bf16* __restrict__ K,
    const __bf16* __restrict__ VT, float* __restrict__ Out,
    int Sq, int SkvPad, int kv_len, float scale)
{
  __shared__ __align__(16) __bf16 ldsK[FKV][KPITCH];
  __shared__ __align__(16) __bf16 ldsVT[128][VPITCH];
  __shared__ __align__(16) __bf16 ldsP[4][16][PPITCH];

  const int h    = blockIdx.y;
  const int tid  = threadIdx.x;
  const int wave = tid >> 5, lane = tid & 31;
  const int half = lane >> 4, l16 = lane & 15;
  const int q0 = blockIdx.x * FQ_TILE + wave * 16;

  // ---- Q fragments (held in VGPRs): 4 d-steps of 32 ----
  v16bf qf[4];
  {
    const __bf16* qrow = Q + ((size_t)h * Sq + q0 + l16) * D_;
#pragma unroll
    for (int ds = 0; ds < 4; ++ds) {
      const int base = ds * 32 + half * 8;
      F16x16 f;
      f.u[0] = *reinterpret_cast<const uint4*>(qrow + base);
      f.u[1] = *reinterpret_cast<const uint4*>(qrow + base + 16);
      qf[ds] = f.v;
    }
  }

  v8f accO[8];
#pragma unroll
  for (int i = 0; i < 8; ++i) accO[i] = v8f_zero();
  float mrow[8], lrow[8];
#pragma unroll
  for (int r = 0; r < 8; ++r) { mrow[r] = -1e30f; lrow[r] = 0.f; }

  const int kr = tid >> 2, kc = (tid & 3) * 32;  // K tile: 32x128, 64B/thread

  for (int kb = 0; kb < SkvPad; kb += FKV) {
    __syncthreads();
    // ---- async-stage K tile [32 keys][128 d] ----
    async_copy64(K + ((size_t)h * SkvPad + kb + kr) * D_ + kc, &ldsK[kr][kc]);
    // ---- async-stage V^T tile [128 d][32 keys] ----
    async_copy64(VT + ((size_t)h * D_ + tid) * SkvPad + kb, &ldsVT[tid][0]);
    async_wait0();
    __syncthreads();

    // ---- scores: S(16x32) = Q(16x128) @ K^T(128x32) ----
    v8f s0 = v8f_zero(), s1 = v8f_zero();
#pragma unroll
    for (int ds = 0; ds < 4; ++ds) {
      F16x16 f0, f1;
      f0.u[0] = *reinterpret_cast<const uint4*>(&ldsK[l16][ds * 32 + half * 8]);
      f0.u[1] = *reinterpret_cast<const uint4*>(&ldsK[l16][ds * 32 + 16 + half * 8]);
      f1.u[0] = *reinterpret_cast<const uint4*>(&ldsK[16 + l16][ds * 32 + half * 8]);
      f1.u[1] = *reinterpret_cast<const uint4*>(&ldsK[16 + l16][ds * 32 + 16 + half * 8]);
      s0 = __builtin_amdgcn_wmma_f32_16x16x32_bf16(false, qf[ds], false, f0.v, (short)0, s0, false, false);
      s1 = __builtin_amdgcn_wmma_f32_16x16x32_bf16(false, qf[ds], false, f1.v, (short)0, s1, false, false);
    }

    // ---- online softmax per row (row = half*8 + r) ----
#pragma unroll
    for (int r = 0; r < 8; ++r) {
      float v0 = s0[r] * scale, v1 = s1[r] * scale;
      if (kb + l16 >= kv_len)      v0 = -1e30f;
      if (kb + 16 + l16 >= kv_len) v1 = -1e30f;
      const float mx   = red_max16(fmaxf(v0, v1));
      const float mnew = fmaxf(mrow[r], mx);
      const float corr = __expf(mrow[r] - mnew);
      mrow[r] = mnew;
      const float p0 = __expf(v0 - mnew);
      const float p1 = __expf(v1 - mnew);
      lrow[r] = lrow[r] * corr + red_sum16(p0 + p1);
#pragma unroll
      for (int dt = 0; dt < 8; ++dt) accO[dt][r] *= corr;
      ldsP[wave][half * 8 + r][l16]      = (__bf16)p0;
      ldsP[wave][half * 8 + r][16 + l16] = (__bf16)p1;
    }
    __syncthreads();

    // ---- O += P(16x32) @ V(32x128) ----
    v16bf pf;
    {
      F16x16 f;
      f.u[0] = *reinterpret_cast<const uint4*>(&ldsP[wave][l16][half * 8]);
      f.u[1] = *reinterpret_cast<const uint4*>(&ldsP[wave][l16][16 + half * 8]);
      pf = f.v;
    }
#pragma unroll
    for (int dt = 0; dt < 8; ++dt) {
      F16x16 f;
      f.u[0] = *reinterpret_cast<const uint4*>(&ldsVT[dt * 16 + l16][half * 8]);
      f.u[1] = *reinterpret_cast<const uint4*>(&ldsVT[dt * 16 + l16][16 + half * 8]);
      accO[dt] = __builtin_amdgcn_wmma_f32_16x16x32_bf16(false, pf, false, f.v, (short)0, accO[dt], false, false);
    }
  }

  // ---- normalize + write ----
#pragma unroll
  for (int dt = 0; dt < 8; ++dt) {
#pragma unroll
    for (int r = 0; r < 8; ++r) {
      const int qi  = q0 + half * 8 + r;
      const int col = h * D_ + dt * 16 + l16;
      Out[(size_t)qi * HID_ + col] = accO[dt][r] / lrow[r];
    }
  }
}

// =====================================================================
// Epilogue builders: one wave per (head, seq) row, lane owns 4 d's.
// Outputs: qf/kf row-major [H][S][D]; V transposed -> [H][D][S].
// =====================================================================
__global__ __launch_bounds__(256) void build_main_kernel(
    const float* __restrict__ q, const float* __restrict__ k, const float* __restrict__ v,
    const float* __restrict__ eq, const float* __restrict__ ek, const float* __restrict__ ev,
    const float* __restrict__ nqw, const float* __restrict__ nkw,
    const float* __restrict__ naqw, const float* __restrict__ nakw,
    const float* __restrict__ cosT, const float* __restrict__ sinT,
    __bf16* __restrict__ qf, __bf16* __restrict__ kf, __bf16* __restrict__ vT)
{
  const int gw = blockIdx.x * 8 + (threadIdx.x >> 5);
  if (gw >= H_ * S_ALL_) return;
  const int h = gw / S_ALL_, s = gw % S_ALL_;
  const int lane = threadIdx.x & 31, d0 = lane * 4;
  const bool txt = (s < S_TXT_);
  const size_t roff = txt ? (size_t)s * HID_ : (size_t)(s - S_TXT_) * HID_;
  const size_t coff = roff + h * D_ + d0;

  const float4 xq = *reinterpret_cast<const float4*>((txt ? eq : q) + coff);
  const float4 xk = *reinterpret_cast<const float4*>((txt ? ek : k) + coff);
  const float4 xv = *reinterpret_cast<const float4*>((txt ? ev : v) + coff);

  float sq = xq.x*xq.x + xq.y*xq.y + xq.z*xq.z + xq.w*xq.w;
  float sk = xk.x*xk.x + xk.y*xk.y + xk.z*xk.z + xk.w*xk.w;
  sq = red_sum32(sq) * (1.f / D_);
  sk = red_sum32(sk) * (1.f / D_);
  const float rq = rsqrtf(sq + EPS_), rk = rsqrtf(sk + EPS_);

  const float* wq = txt ? naqw : nqw;
  const float* wk = txt ? nakw : nkw;
  float nq[4] = { xq.x*rq*wq[d0], xq.y*rq*wq[d0+1], xq.z*rq*wq[d0+2], xq.w*rq*wq[d0+3] };
  float nk[4] = { xk.x*rk*wk[d0], xk.y*rk*wk[d0+1], xk.z*rk*wk[d0+2], xk.w*rk*wk[d0+3] };

  const float4 cc = *reinterpret_cast<const float4*>(cosT + (size_t)s * D_ + d0);
  const float4 ss = *reinterpret_cast<const float4*>(sinT + (size_t)s * D_ + d0);
  float oq[4], ok[4];
  oq[0] = nq[0]*cc.x - nq[1]*ss.x;  oq[1] = nq[1]*cc.y + nq[0]*ss.y;
  oq[2] = nq[2]*cc.z - nq[3]*ss.z;  oq[3] = nq[3]*cc.w + nq[2]*ss.w;
  ok[0] = nk[0]*cc.x - nk[1]*ss.x;  ok[1] = nk[1]*cc.y + nk[0]*ss.y;
  ok[2] = nk[2]*cc.z - nk[3]*ss.z;  ok[3] = nk[3]*cc.w + nk[2]*ss.w;

  const size_t qoff = ((size_t)h * S_ALL_ + s) * D_ + d0;
  const float xvv[4] = { xv.x, xv.y, xv.z, xv.w };
#pragma unroll
  for (int j = 0; j < 4; ++j) {
    qf[qoff + j] = (__bf16)oq[j];
    kf[qoff + j] = (__bf16)ok[j];
    vT[((size_t)h * D_ + d0 + j) * S_ALL_ + s] = (__bf16)xvv[j];
  }
}

__global__ __launch_bounds__(256) void build_ipq_kernel(
    const float* __restrict__ q, const float* __restrict__ nipq,
    __bf16* __restrict__ ipq)
{
  const int gw = blockIdx.x * 8 + (threadIdx.x >> 5);
  if (gw >= H_ * S_IMG_) return;
  const int h = gw / S_IMG_, s = gw % S_IMG_;
  const int lane = threadIdx.x & 31, d0 = lane * 4;
  const float4 x = *reinterpret_cast<const float4*>(q + (size_t)s * HID_ + h * D_ + d0);
  float ms = red_sum32(x.x*x.x + x.y*x.y + x.z*x.z + x.w*x.w) * (1.f / D_);
  const float r = rsqrtf(ms + EPS_);
  const float xv[4] = { x.x, x.y, x.z, x.w };
  const size_t o = ((size_t)h * S_IMG_ + s) * D_ + d0;
#pragma unroll
  for (int j = 0; j < 4; ++j) ipq[o + j] = (__bf16)(xv[j] * r * nipq[d0 + j]);
}

__global__ __launch_bounds__(256) void build_ipkv_kernel(
    const float* __restrict__ ipk, const float* __restrict__ ipv,
    const float* __restrict__ nipk,
    __bf16* __restrict__ ipkf, __bf16* __restrict__ ipvT)
{
  const int gw = blockIdx.x * 8 + (threadIdx.x >> 5);
  if (gw >= H_ * KVP_IP) return;
  const int h = gw / KVP_IP, jr = gw % KVP_IP;
  const int lane = threadIdx.x & 31, d0 = lane * 4;
  float kk[4] = {0.f,0.f,0.f,0.f}, vv[4] = {0.f,0.f,0.f,0.f};
  if (jr < L_IP_) {
    const float4 xk = *reinterpret_cast<const float4*>(ipk + (size_t)jr * HID_ + h * D_ + d0);
    const float4 xv = *reinterpret_cast<const float4*>(ipv + (size_t)jr * HID_ + h * D_ + d0);
    float ms = red_sum32(xk.x*xk.x + xk.y*xk.y + xk.z*xk.z + xk.w*xk.w) * (1.f / D_);
    const float r = rsqrtf(ms + EPS_);
    kk[0] = xk.x*r*nipk[d0];   kk[1] = xk.y*r*nipk[d0+1];
    kk[2] = xk.z*r*nipk[d0+2]; kk[3] = xk.w*r*nipk[d0+3];
    vv[0] = xv.x; vv[1] = xv.y; vv[2] = xv.z; vv[3] = xv.w;
  }
#pragma unroll
  for (int j = 0; j < 4; ++j) {
    ipkf[((size_t)h * KVP_IP + jr) * D_ + d0 + j]  = (__bf16)kk[j];
    ipvT[((size_t)h * D_ + d0 + j) * KVP_IP + jr]  = (__bf16)vv[j];
  }
}

// =====================================================================
// launch
// =====================================================================
extern "C" void kernel_launch(void* const* d_in, const int* in_sizes, int n_in,
                              void* d_out, int out_size, void* d_ws, size_t ws_size,
                              hipStream_t stream) {
  (void)in_sizes; (void)n_in; (void)out_size; (void)ws_size;
  const float* hidden = (const float*)d_in[0];
  const float* enc    = (const float*)d_in[1];
  const float* iph    = (const float*)d_in[2];
  const float* cosT   = (const float*)d_in[3];
  const float* sinT   = (const float*)d_in[4];
  const float* Wq  = (const float*)d_in[5];   const float* bq  = (const float*)d_in[6];
  const float* Wk  = (const float*)d_in[7];   const float* bk  = (const float*)d_in[8];
  const float* Wv  = (const float*)d_in[9];   const float* bv  = (const float*)d_in[10];
  const float* nqw = (const float*)d_in[11];  const float* nkw = (const float*)d_in[12];
  const float* Waq = (const float*)d_in[13];  const float* baq = (const float*)d_in[14];
  const float* Wak = (const float*)d_in[15];  const float* bak = (const float*)d_in[16];
  const float* Wav = (const float*)d_in[17];  const float* bav = (const float*)d_in[18];
  const float* naqw = (const float*)d_in[19]; const float* nakw = (const float*)d_in[20];
  const float* Wo  = (const float*)d_in[21];  const float* bo  = (const float*)d_in[22];
  const float* Wad = (const float*)d_in[23];  const float* bad = (const float*)d_in[24];
  const float* Wkip = (const float*)d_in[25]; const float* bkip = (const float*)d_in[26];
  const float* Wvip = (const float*)d_in[27]; const float* bvip = (const float*)d_in[28];
  const float* nipq = (const float*)d_in[29]; const float* nipk = (const float*)d_in[30];

  char* ws = (char*)d_ws;
  auto alloc = [&](size_t bytes) -> void* {
    void* p = ws; ws += (bytes + 255) & ~(size_t)255; return p;
  };
  float* qb    = (float*)alloc((size_t)S_IMG_ * HID_ * 4);
  float* kb_   = (float*)alloc((size_t)S_IMG_ * HID_ * 4);
  float* vb    = (float*)alloc((size_t)S_IMG_ * HID_ * 4);
  float* eqb   = (float*)alloc((size_t)S_TXT_ * HID_ * 4);
  float* ekb   = (float*)alloc((size_t)S_TXT_ * HID_ * 4);
  float* evb   = (float*)alloc((size_t)S_TXT_ * HID_ * 4);
  float* ipkb  = (float*)alloc((size_t)L_IP_ * HID_ * 4);
  float* ipvb_f= (float*)alloc((size_t)L_IP_ * HID_ * 4);
  float* attn  = (float*)alloc((size_t)S_ALL_ * HID_ * 4);
  float* ipout = (float*)alloc((size_t)S_IMG_ * HID_ * 4);
  __bf16* qfb   = (__bf16*)alloc((size_t)H_ * S_ALL_ * D_ * 2);
  __bf16* kfb   = (__bf16*)alloc((size_t)H_ * S_ALL_ * D_ * 2);
  __bf16* vTb   = (__bf16*)alloc((size_t)H_ * D_ * S_ALL_ * 2);
  __bf16* ipqb  = (__bf16*)alloc((size_t)H_ * S_IMG_ * D_ * 2);
  __bf16* ipkfb = (__bf16*)alloc((size_t)H_ * KVP_IP * D_ * 2);
  __bf16* ipvTb = (__bf16*)alloc((size_t)H_ * D_ * KVP_IP * 2);

  float* outImg = (float*)d_out;                                  // [1024,3072]
  float* outEnc = (float*)d_out + (size_t)S_IMG_ * HID_;          // [512,3072]

  const dim3 gImg(HID_ / BN, S_IMG_ / BM);   // (24,8)
  const dim3 gTxt(HID_ / BN, S_TXT_ / BM);   // (24,4)
  const dim3 gIp (HID_ / BN, 1);

  // qkv projections (image stream)
  gemm_bias_kernel<<<gImg, 256, 0, stream>>>(hidden, nullptr, Wq, bq, qb,  S_IMG_, HID_, HID_);
  gemm_bias_kernel<<<gImg, 256, 0, stream>>>(hidden, nullptr, Wk, bk, kb_, S_IMG_, HID_, HID_);
  gemm_bias_kernel<<<gImg, 256, 0, stream>>>(hidden, nullptr, Wv, bv, vb,  S_IMG_, HID_, HID_);
  // text stream
  gemm_bias_kernel<<<gTxt, 256, 0, stream>>>(enc, nullptr, Waq, baq, eqb, S_TXT_, HID_, HID_);
  gemm_bias_kernel<<<gTxt, 256, 0, stream>>>(enc, nullptr, Wak, bak, ekb, S_TXT_, HID_, HID_);
  gemm_bias_kernel<<<gTxt, 256, 0, stream>>>(enc, nullptr, Wav, bav, evb, S_TXT_, HID_, HID_);
  // IP stream
  gemm_bias_kernel<<<gIp, 256, 0, stream>>>(iph, nullptr, Wkip, bkip, ipkb,   L_IP_, HID_, IPD_);
  gemm_bias_kernel<<<gIp, 256, 0, stream>>>(iph, nullptr, Wvip, bvip, ipvb_f, L_IP_, HID_, IPD_);

  // epilogue builders (RMS + RoPE + head layouts, bf16)
  build_main_kernel<<<(H_ * S_ALL_) / 8, 256, 0, stream>>>(
      qb, kb_, vb, eqb, ekb, evb, nqw, nkw, naqw, nakw, cosT, sinT, qfb, kfb, vTb);
  build_ipq_kernel<<<(H_ * S_IMG_) / 8, 256, 0, stream>>>(qb, nipq, ipqb);
  build_ipkv_kernel<<<(H_ * KVP_IP) / 8, 256, 0, stream>>>(ipkb, ipvb_f, nipk, ipkfb, ipvTb);

  const float scale = 0.08838834764831845f;  // 1/sqrt(128)
  // main joint attention over S=1536
  flash_attn_kernel<<<dim3(S_ALL_ / FQ_TILE, H_), 128, 0, stream>>>(
      qfb, kfb, vTb, attn, S_ALL_, S_ALL_, S_ALL_, scale);
  // IP attention (kv padded 16 -> 32, masked)
  flash_attn_kernel<<<dim3(S_IMG_ / FQ_TILE, H_), 128, 0, stream>>>(
      ipqb, ipkfb, ipvTb, ipout, S_IMG_, KVP_IP, L_IP_, scale);

  // output projections (img fuses + ip_out)
  gemm_bias_kernel<<<gImg, 256, 0, stream>>>(attn + (size_t)S_TXT_ * HID_, ipout,
                                             Wo, bo, outImg, S_IMG_, HID_, HID_);
  gemm_bias_kernel<<<gTxt, 256, 0, stream>>>(attn, nullptr, Wad, bad, outEnc, S_TXT_, HID_, HID_);
}
